// SelfAttention_47519518163202
// MI455X (gfx1250) — matrix operand order repaired and checked by reading, not compiled, not run on hardware
//
#include <hip/hip_runtime.h>
#include <hip/hip_bf16.h>

#define B_  2
#define S_  4096
#define H_  8
#define DH_ 64
#define D_  512
#define M_  (B_ * S_)   // 8192

typedef __attribute__((ext_vector_type(16))) __bf16 bf16x16;
typedef __attribute__((ext_vector_type(8)))  __bf16 bf16x8;
typedef __attribute__((ext_vector_type(8)))  float  f32x8;
typedef __attribute__((ext_vector_type(4)))  float  f32x4;

// native f32 -> bf16 (RNE) — lowers to v_cvt_pk_bf16_f32
static __device__ __forceinline__ __bf16 f2bf(float f) { return (__bf16)f; }

static __device__ __forceinline__ bf16x16 joinv(bf16x8 lo, bf16x8 hi) {
    union { bf16x16 v; bf16x8 h[2]; } u;
    u.h[0] = lo; u.h[1] = hi;
    return u.v;
}

// convert 8 contiguous fp32 to bf16x8
static __device__ __forceinline__ bf16x8 cvt8(const float* __restrict__ p) {
    f32x4 a = *(const f32x4*)p;
    f32x4 b = *(const f32x4*)(p + 4);
    bf16x8 r;
#pragma unroll
    for (int i = 0; i < 4; ++i) { r[i] = f2bf(a[i]); r[i + 4] = f2bf(b[i]); }
    return r;
}

static __device__ __forceinline__ f32x8 wmma_bf16(bf16x16 a, bf16x16 b, f32x8 c) {
    return __builtin_amdgcn_wmma_f32_16x16x32_bf16(false, a, false, b, (short)0, c,
                                                   false, false);
}

// one async 16-byte global->LDS copy (ASYNCcnt-tracked)
static __device__ __forceinline__ void async_cp16(unsigned lds_off,
                                                  unsigned long long gaddr) {
    asm volatile("global_load_async_to_lds_b128 %0, %1, off"
                 :: "v"(lds_off), "v"(gaddr) : "memory");
}

// ---------------------------------------------------------------------------
// Kernel 1: Q/K/V projection.  Y = X @ W^T  (X fp32 [M,512], W fp32 [512,512])
// grid.y: 0->Q ([B,H,S,DH]), 1->K ([B,H,S,DH]), 2->V transposed ([B,H,DH,S])
// ---------------------------------------------------------------------------
__global__ __launch_bounds__(32)
void qkv_proj_kernel(const float* __restrict__ X,
                     const float* __restrict__ Wq,
                     const float* __restrict__ Wk,
                     const float* __restrict__ Wv,
                     __bf16* __restrict__ Qb,
                     __bf16* __restrict__ Kb,
                     __bf16* __restrict__ Vt) {
    const int lane = threadIdx.x;
    const int ni = lane & 15;       // A-row / B-col index within tile
    const int hg = lane >> 4;       // lane half-group
    const int which = blockIdx.y;
    const float* W = (which == 0) ? Wq : (which == 1) ? Wk : Wv;

    const int tilesM = M_ / 16;     // 512
    const int m0 = (blockIdx.x % tilesM) * 16;
    const int n0 = (blockIdx.x / tilesM) * 16;

    const float* arow = X + (size_t)(m0 + ni) * D_;
    const float* brow = W + (size_t)(n0 + ni) * D_;

    f32x8 c = {};
    for (int kk = 0; kk < D_; kk += 32) {
        bf16x16 a = joinv(cvt8(arow + kk + hg * 16),
                          cvt8(arow + kk + hg * 16 + 8));
        bf16x16 b = joinv(cvt8(brow + kk + hg * 16),
                          cvt8(brow + kk + hg * 16 + 8));
        c = wmma_bf16(a, b, c);
    }

    // epilogue: tile rows never straddle a batch boundary (m0 % 16 == 0)
    const int ncol = n0 + ni;
    const int hh = ncol >> 6;                 // head
    const int dh = ncol & 63;
    const int bb = m0 >> 12;                  // m0 / S_
    const int s0 = (m0 & (S_ - 1)) + 8 * hg;  // first seq pos of this lane's rows

    if (which == 2) {
        // V transposed [B,H,DH,S]: 8 consecutive seq positions -> one b128 store
        bf16x8 pack;
#pragma unroll
        for (int j = 0; j < 8; ++j) pack[j] = f2bf(c[j]);
        *(bf16x8*)(Vt + (((size_t)bb * H_ + hh) * DH_ + dh) * S_ + s0) = pack;
    } else {
        __bf16* dst = (which == 0) ? Qb : Kb;
        __bf16* base = dst + (((size_t)bb * H_ + hh) * S_ + s0) * DH_ + dh;
#pragma unroll
        for (int j = 0; j < 8; ++j) base[(size_t)j * DH_] = f2bf(c[j]);
    }
}

// ---------------------------------------------------------------------------
// Kernel 2: flash attention.  One wave per (batch, head, 16-query tile).
// K/V tiles double-buffered in LDS via global_load_async_to_lds_b128.
// ---------------------------------------------------------------------------
__global__ __launch_bounds__(32)
void flash_attn_kernel(const __bf16* __restrict__ Qb,
                       const __bf16* __restrict__ Kb,
                       const __bf16* __restrict__ Vt,
                       __bf16* __restrict__ Hb) {
    __shared__ __bf16 Kbuf[2][32 * DH_];    // [key][dh]   4KB per buffer
    __shared__ __bf16 Vbuf[2][DH_ * 32];    // [d][key]    4KB per buffer
    __shared__ __bf16 Plds[16 * 32];        // P transpose staging

    const int lane = threadIdx.x;
    const int ni = lane & 15;
    const int hg = lane >> 4;
    const int qt = blockIdx.x;              // query tile (0..255)
    const int hd = blockIdx.y;              // head
    const int bz = blockIdx.z;              // batch

    const __bf16* Qp = Qb + (((size_t)bz * H_ + hd) * S_ + (size_t)qt * 16) * DH_;
    const __bf16* Kp = Kb + ((size_t)bz * H_ + hd) * S_ * DH_;
    const __bf16* Vp = Vt + ((size_t)bz * H_ + hd) * DH_ * S_;

    // issue async copies of the (kt..kt+31) K/V tiles into buffer `buf`
    auto copy_tiles = [&](int kt, int buf) {
        // K tile: 32 rows x 128B, fully contiguous (4096 B)
        unsigned long long ksrc =
            (unsigned long long)(uintptr_t)((const char*)Kp + (size_t)kt * (DH_ * 2));
        unsigned kdst = (unsigned)(uintptr_t)&Kbuf[buf][0];
#pragma unroll
        for (int i = 0; i < 8; ++i) {
            unsigned off = (unsigned)(lane + 32 * i) * 16u;
            async_cp16(kdst + off, ksrc + off);
        }
        // V tile: 64 rows x 64B, row stride S_*2 bytes
        unsigned long long vsrc =
            (unsigned long long)(uintptr_t)((const char*)Vp + (size_t)kt * 2);
        unsigned vdst = (unsigned)(uintptr_t)&Vbuf[buf][0];
#pragma unroll
        for (int i = 0; i < 8; ++i) {
            int c = lane + 32 * i;          // 0..255
            int row = c >> 2;               // 0..63
            int cc = c & 3;                 // 16B chunk within row
            async_cp16(vdst + (unsigned)(row * 64 + cc * 16),
                       vsrc + (size_t)row * (S_ * 2) + (size_t)cc * 16);
        }
    };

    // Q A-fragments (kdim 0..31 and 32..63), loaded once from global
    const __bf16* qrow = Qp + (size_t)ni * DH_;
    bf16x16 aQ0 = joinv(*(const bf16x8*)(qrow + hg * 16),
                        *(const bf16x8*)(qrow + hg * 16 + 8));
    bf16x16 aQ1 = joinv(*(const bf16x8*)(qrow + 32 + hg * 16),
                        *(const bf16x8*)(qrow + 32 + hg * 16 + 8));

    f32x8 o0 = {}, o1 = {}, o2 = {}, o3 = {};
    float mrow[8], lrow[8];
#pragma unroll
    for (int j = 0; j < 8; ++j) { mrow[j] = -1e30f; lrow[j] = 0.0f; }

    copy_tiles(0, 0);
    asm volatile("s_wait_asynccnt 0" ::: "memory");

    for (int kt = 0; kt < S_; kt += 32) {
        const int buf = (kt >> 5) & 1;
        if (kt + 32 < S_) copy_tiles(kt + 32, buf ^ 1);   // prefetch next tile

        // scores: S = Q @ K^T (two 16-key sub-tiles, kdim 64 = 2 WMMAs each)
        f32x8 s0 = {}, s1 = {};
        {
            const __bf16* krow = &Kbuf[buf][(size_t)ni * DH_];
            bf16x16 b0 = joinv(*(const bf16x8*)(krow + hg * 16),
                               *(const bf16x8*)(krow + hg * 16 + 8));
            bf16x16 b1 = joinv(*(const bf16x8*)(krow + 32 + hg * 16),
                               *(const bf16x8*)(krow + 32 + hg * 16 + 8));
            s0 = wmma_bf16(aQ0, b0, s0);
            s0 = wmma_bf16(aQ1, b1, s0);
        }
        {
            const __bf16* krow = &Kbuf[buf][(size_t)(16 + ni) * DH_];
            bf16x16 b0 = joinv(*(const bf16x8*)(krow + hg * 16),
                               *(const bf16x8*)(krow + hg * 16 + 8));
            bf16x16 b1 = joinv(*(const bf16x8*)(krow + 32 + hg * 16),
                               *(const bf16x8*)(krow + 32 + hg * 16 + 8));
            s1 = wmma_bf16(aQ0, b0, s1);
            s1 = wmma_bf16(aQ1, b1, s1);
        }

        // online softmax over this 32-key tile (rows r = j + 8*hg)
#pragma unroll
        for (int j = 0; j < 8; ++j) {
            float a = s0[j] * 0.125f;
            float b = s1[j] * 0.125f;
            float t = fmaxf(a, b);
            t = fmaxf(t, __shfl_xor(t, 1, 32));
            t = fmaxf(t, __shfl_xor(t, 2, 32));
            t = fmaxf(t, __shfl_xor(t, 4, 32));
            t = fmaxf(t, __shfl_xor(t, 8, 32));
            float mnew = fmaxf(mrow[j], t);
            float rs = __expf(mrow[j] - mnew);
            float p0 = __expf(a - mnew);
            float p1 = __expf(b - mnew);
            float rsum = p0 + p1;
            rsum += __shfl_xor(rsum, 1, 32);
            rsum += __shfl_xor(rsum, 2, 32);
            rsum += __shfl_xor(rsum, 4, 32);
            rsum += __shfl_xor(rsum, 8, 32);
            lrow[j] = lrow[j] * rs + rsum;
            mrow[j] = mnew;
            o0[j] *= rs; o1[j] *= rs; o2[j] *= rs; o3[j] *= rs;
            int r = j + 8 * hg;
            Plds[r * 32 + ni]      = f2bf(p0);
            Plds[r * 32 + 16 + ni] = f2bf(p1);
        }
        asm volatile("s_wait_dscnt 0" ::: "memory");

        // transpose P into A-fragment layout via LDS
    bf16x16 aP = joinv(*(const bf16x8*)&Plds[ni * 32 + hg * 16],
                           *(const bf16x8*)&Plds[ni * 32 + hg * 16 + 8]);

        // attn += P @ V  (V tile in LDS, contiguous along keys)
#pragma unroll
        for (int t = 0; t < 4; ++t) {
            const __bf16* vrow = &Vbuf[buf][(size_t)(t * 16 + ni) * 32 + hg * 16];
            bf16x16 bV = joinv(*(const bf16x8*)vrow, *(const bf16x8*)(vrow + 8));
            f32x8& o = (t == 0) ? o0 : (t == 1) ? o1 : (t == 2) ? o2 : o3;
            o = wmma_bf16(aP, bV, o);
        }

        // next iteration reads buf^1: its async copies must have landed
        asm volatile("s_wait_asynccnt 0" ::: "memory");
    }

    // epilogue: normalize and write h_ [B,S,D] bf16
#pragma unroll
    for (int j = 0; j < 8; ++j) {
        int r = j + 8 * hg;
        int s = qt * 16 + r;
        float inv = 1.0f / lrow[j];
        size_t base = ((size_t)bz * S_ + s) * D_ + (size_t)hd * DH_;
        Hb[base + 0 * 16 + ni] = f2bf(o0[j] * inv);
        Hb[base + 1 * 16 + ni] = f2bf(o1[j] * inv);
        Hb[base + 2 * 16 + ni] = f2bf(o2[j] * inv);
        Hb[base + 3 * 16 + ni] = f2bf(o3[j] * inv);
    }
}

// ---------------------------------------------------------------------------
// Kernel 3: output projection.  Out = h_ @ Wo^T + bo  (f32 output)
// ---------------------------------------------------------------------------
__global__ __launch_bounds__(32)
void out_proj_kernel(const __bf16* __restrict__ Hb,
                     const float* __restrict__ Wo,
                     const float* __restrict__ bo,
                     float* __restrict__ Out) {
    const int lane = threadIdx.x;
    const int ni = lane & 15;
    const int hg = lane >> 4;
    const int tilesM = M_ / 16;
    const int m0 = (blockIdx.x % tilesM) * 16;
    const int n0 = (blockIdx.x / tilesM) * 16;

    const __bf16* arow = Hb + (size_t)(m0 + ni) * D_;
    const float*  brow = Wo + (size_t)(n0 + ni) * D_;

    f32x8 c = {};
    for (int kk = 0; kk < D_; kk += 32) {
        bf16x16 a = joinv(*(const bf16x8*)(arow + kk + hg * 16),
                          *(const bf16x8*)(arow + kk + hg * 16 + 8));
        bf16x16 b = joinv(cvt8(brow + kk + hg * 16),
                          cvt8(brow + kk + hg * 16 + 8));
        c = wmma_bf16(a, b, c);
    }
    float bias = bo[n0 + ni];
    float* obase = Out + (size_t)(m0 + 8 * hg) * D_ + n0 + ni;
#pragma unroll
    for (int j = 0; j < 8; ++j) {
        obase[(size_t)j * D_] = c[j] + bias;
    }
}

extern "C" void kernel_launch(void* const* d_in, const int* in_sizes, int n_in,
                              void* d_out, int out_size, void* d_ws, size_t ws_size,
                              hipStream_t stream) {
    const float* x  = (const float*)d_in[0];
    const float* Wq = (const float*)d_in[1];
    const float* Wk = (const float*)d_in[2];
    const float* Wv = (const float*)d_in[3];
    const float* Wo = (const float*)d_in[4];
    const float* bo = (const float*)d_in[5];
    float* out = (float*)d_out;

    char* ws = (char*)d_ws;
    const size_t SZ = (size_t)B_ * H_ * S_ * DH_ * sizeof(__bf16);  // 8 MB
    __bf16* Qb = (__bf16*)(ws);
    __bf16* Kb = (__bf16*)(ws + SZ);
    __bf16* Vt = (__bf16*)(ws + 2 * SZ);
    __bf16* Hb = (__bf16*)(ws + 3 * SZ);

    dim3 g1((M_ / 16) * (D_ / 16), 3);              // 16384 tiles x {Q,K,V}
    qkv_proj_kernel<<<g1, 32, 0, stream>>>(x, Wq, Wk, Wv, Qb, Kb, Vt);

    dim3 g2(S_ / 16, H_, B_);                       // 256 x 8 x 2 waves
    flash_attn_kernel<<<g2, 32, 0, stream>>>(Qb, Kb, Vt, Hb);

    out_proj_kernel<<<(M_ / 16) * (D_ / 16), 32, 0, stream>>>(Hb, Wo, bo, out);
}
